// complex_msl_backbone_72739566125505
// MI455X (gfx1250) — matrix operand order repaired and checked by reading, not compiled, not run on hardware
//
#include <hip/hip_runtime.h>

typedef __bf16 bf16_t;
typedef __attribute__((ext_vector_type(16))) __bf16 v16bf;
typedef __attribute__((ext_vector_type(8)))  float  v8f;

#define NTHR   256
#define NWAVES 8

// ---------------- LDS arena (byte offsets), peak 323264 B <= 320KB/WGP ----------------
#define X2_OFS      0        // (125,3,10) stride 3750  (alive phase B..C)
#define X1_OFS      120000   // (25,5,18)  stride 2250  (alive phase A..B)
#define XIN_OFS     192000   // (1,5,27)   stride 135
#define BUFA_OFS    196320   // (5,5,27)   stride 675
#define BUFB_OFS    217920   // (5,5,18)   stride 450
#define BUFC_OFS    232320   // (1,5,18)   stride 90
#define Z1ROW_OFS   192000   // (75,1,18)  stride 1350  (phase B, reuses XIN/BUFA)
#define SL1R_OFS    235200   // (75,2,10)  stride 1500  rolling 2-row buffer
#define SLRA_OFS    283200   // (25,5,10)  stride 1250
#define SLR02_OFS   120000   // (125,3,5)  stride 1875  (phase C, reuses X1)
#define W0ROW_OFS   120000   // (300,1,10) stride 3000  (phase C, after SLR02 consumed)
#define SL2_OFS     216000   // (300,2,5)  stride 3000
#define ZSLOT_OFS   323200   // 2-byte zero slot backing branchless pad gathers
#define SMEM_BYTES  323264

// ---------------- weight fragment offsets in d_ws (bf16 elements) ----------------
#define FR_C0A   0
#define FR_SL0   512
#define FR_W0B   1536
#define FR_WR0   2560
#define FR_W0C   3584
#define FR_WRA   4608
#define FR_WA    5120
#define FR_W1    17408
#define FR_WL1   22528
#define FR_W2    23040
#define FR_WR02  43520
#define FR_W0C2  44032
#define FR_W0    240640
#define FR_WL2   318464
#define FR_W0B2  318976
// total 630272 elems = 1.26 MB in d_ws (L2-resident, shared by all 1024 WGs)

// Generic 16(batch) x N x K GEMM engine: bf16 WMMA, f32 accumulate.
// All geometry is compile-time so address spaces infer (ds_* / global_*),
// divisions fold, and the pad checks become cndmask selects.
template<int TYPE, int CIN, int HINBUF, int HINREAL, int WIN, int WRAP,
         int HOUTBUF, int WOUT, int NV, int KV, int PH, int HPOS, int FLAGS,
         int INSTRIDE, int OUTSTRIDE, int NPOS, int INOFS, int OUTOFS>
__device__ __attribute__((always_inline)) inline void run_gemm(
    const bf16_t* __restrict__ Bfrag, const float* __restrict__ bias,
    float* __restrict__ gout, int hoBase, int hOutOfs)
{
  extern __shared__ char smem[];
  const bf16_t* lds  = (const bf16_t*)smem;
  bf16_t*       ldsw = (bf16_t*)smem;

  constexpr int KT = (KV + 31) / 32;
  constexpr int NT = (NV + 15) / 16;
  constexpr int ZEL = ZSLOT_OFS / 2;

  __syncthreads();   // producer->consumer + WAR on arena reuse

  const int tid  = threadIdx.x;
  const int lane = tid & 31;
  const int wave = tid >> 5;
  const int m    = lane & 15;                 // batch row within 16-sample tile
  const int kl0  = (lane >> 4) << 3;          // A frag: lanes16-31 start at K=8
  const int inEl = INOFS / 2 + m * INSTRIDE;

  for (int t = wave; t < NPOS * NT; t += NWAVES) {
    const int ntile = t % NT;
    const int pos   = t / NT;
    int ho = 0, w = 0, c = 0, hh = 0, rowBase = 0;
    if (TYPE == 0) { ho = hoBase + pos / WOUT; w = pos % WOUT; }
    else           { c = pos / HPOS; hh = pos - c * HPOS;
                     rowBase = (c * HINBUF + hh) * WIN; }

    const bf16_t* bptr = Bfrag + (size_t)(ntile * KT) * 512 + lane * 16;
    v8f acc = {0.f,0.f,0.f,0.f,0.f,0.f,0.f,0.f};

    for (int kt = 0; kt < KT; ++kt) {
      v16bf a;
#pragma unroll
      for (int j = 0; j < 16; ++j) {
        // ISA 16-bit A layout: lanes0-15 K=0-7,16-23 ; lanes16-31 K=8-15,24-31
        const int k = kt * 32 + kl0 + (j & 7) + ((j >> 3) << 4);
        bool valid = (k < KV);
        int  off;
        if (TYPE == 0) {
          const int dh = k / CIN;              // constant divisor -> mul/shift
          const int ic = k - dh * CIN;
          const int hi = ho + dh - PH;
          valid = valid && (hi >= 0) && (hi < HINREAL);
          const int slot = WRAP ? (hi & 1) : hi;
          off = inEl + ic * (HINBUF * WIN) + slot * WIN + w;
        } else {
          off = inEl + rowBase + k;
        }
        off = valid ? off : ZEL;               // branchless: pad -> zero slot
        a[j] = lds[off];                       // ds_load_u16
      }
      v16bf b = *(const v16bf*)(bptr + kt * 512);   // 2x global_load_b128 (L2 hot)
      acc = __builtin_amdgcn_wmma_f32_16x16x32_bf16(
              false, a, false, b, (short)0, acc, false, false);
    }

    const int n = ntile * 16 + (lane & 15);
    if (n < NV) {
      const float bb  = bias[n];
      const int   mof = (lane >> 4) << 3;      // C/D: lanes16-31 hold M+8
#pragma unroll
      for (int r = 0; r < 8; ++r) {
        const int mm = r + mof;
        float v = acc[r] + bb;
        if (FLAGS & 1) v = v > 0.f ? v : 0.f;
        if (FLAGS & 4) {                        // (B,512,1,5) f32 global output
          size_t go = (size_t)mm * 2560 + (size_t)n * WOUT + w;
          if (FLAGS & 2) gout[go] += v; else gout[go] = v;
        } else {
          int off;
          if (TYPE == 0) { const int hs = ho % HOUTBUF;
                           off = (n * HOUTBUF + hs) * WOUT + w; }
          else           { const int hs = (hh + hOutOfs) % HOUTBUF;
                           off = (c * HOUTBUF + hs) * WOUT + n; }
          bf16_t* dst = ldsw + OUTOFS / 2 + mm * OUTSTRIDE + off;
          if (FLAGS & 2) v += (float)*dst;      // residual add in LDS
          *dst = (bf16_t)v;
        }
      }
    }
  }
}

// Preformat weights into WMMA B-fragment lane order (K x N, zero padded).
__global__ void prep_frag(const float* __restrict__ W, bf16_t* __restrict__ dst,
                          int type, int Cin, int kh, int K, int N, int ktiles, int ntiles) {
  int idx = blockIdx.x * NTHR + threadIdx.x;
  int total = ntiles * ktiles * 512;
  if (idx >= total) return;
  int blk = idx >> 9, r = idx & 511;
  int ntile = blk / ktiles, ktile = blk - ntile * ktiles;
  int lane = r >> 4, j = r & 15;
  int k = ktile * 32 + ((lane >> 4) << 3) + (j & 7) + ((j >> 3) << 4);   // B: lanes0-15 K=0-7,16-23 ; lanes16-31 K=8-15,24-31
  int n = ntile * 16 + (lane & 15);              //    lane&15 = N column
  float v = 0.f;
  if (k < K && n < N) {
    if (type == 0) { int dh = k / Cin, ic = k - dh * Cin; v = W[(n * Cin + ic) * kh + dh]; }
    else           { v = W[n * K + k]; }
  }
  dst[idx] = (bf16_t)v;
}

__global__ __launch_bounds__(NTHR) void msl_main(
    const float* __restrict__ x, const bf16_t* __restrict__ ws, float* __restrict__ out,
    const float* b0a, const float* bl0, const float* b0b, const float* br0, const float* b0c,
    const float* bra, const float* ba,  const float* b1,  const float* bl1, const float* b2,
    const float* br02, const float* b0c2, const float* b0, const float* bl2, const float* b0b2) {
  extern __shared__ char smem[];
  bf16_t* ldsw = (bf16_t*)smem;

  const int tid = threadIdx.x;
  const size_t base = (size_t)blockIdx.x * 16;
  float* gout = out + base * 2560;

  if (tid == 0) ldsw[ZSLOT_OFS / 2] = (bf16_t)0.f;
  // stage inputs: 16 samples of x (1,5,27) f32 -> bf16 LDS
  for (int i = tid; i < 16 * 135; i += NTHR) {
    int s = i / 135, e = i - s * 135;
    ldsw[XIN_OFS / 2 + s * 135 + e] = (bf16_t)x[(base + s) * 135 + e];
  }

  // ---- stage 0 ----
  //        TYPE CIN HB HR WIN WR HOB WOUT  NV  KV PH HP FL  IS   OS   NPOS  INOFS     OUTOFS
  run_gemm<0,  1, 5, 5, 27, 0,  5, 27,   5,  3, 1, 1, 0, 135, 675, 135, XIN_OFS,  BUFA_OFS>(ws+FR_C0A, b0a, nullptr, 0, 0);
  run_gemm<1,  1, 5, 5, 27, 0,  5, 18,  18, 27, 0, 5, 1, 675, 450,  25, BUFA_OFS, BUFB_OFS>(ws+FR_SL0, bl0, nullptr, 0, 0);   // relu
  run_gemm<0,  5, 5, 5, 18, 0,  5, 18,  25, 15, 1, 1, 0, 450, 2250, 90, BUFB_OFS, X1_OFS  >(ws+FR_W0B, b0b, nullptr, 0, 0);
  run_gemm<1,  1, 5, 5, 27, 0,  5, 18,  18, 27, 0, 5, 0, 135,  90,   5, XIN_OFS,  BUFC_OFS>(ws+FR_WR0, br0, nullptr, 0, 0);
  run_gemm<0,  1, 5, 5, 18, 0,  5, 18,  25,  1, 0, 1, 2,  90, 2250, 90, BUFC_OFS, X1_OFS  >(ws+FR_W0C, b0c, nullptr, 0, 0);   // += (x1=z+y)

  // ---- stage 1: residual path first, then per-row fused main path ----
  run_gemm<1,  1, 5, 5, 18, 0,  5, 10,  10, 18, 0, 5, 0, 2250,1250,125, X1_OFS,   SLRA_OFS>(ws+FR_WRA, bra, nullptr, 0, 0);
  run_gemm<0, 25, 5, 5, 10, 0,  3, 10, 125, 75, 0, 1, 0, 1250,3750, 30, SLRA_OFS, X2_OFS  >(ws+FR_WA,  ba,  nullptr, 0, 0);
#pragma unroll 1
  for (int h = 0; h < 4; ++h) {
    run_gemm<0, 25, 5, 5, 18, 0, 1, 18,  75, 50, 0, 1, 0, 2250,1350, 18, X1_OFS,   Z1ROW_OFS>(ws+FR_W1, b1, nullptr, h, 0);
    run_gemm<1,  1, 1, 1, 18, 0, 2, 10,  10, 18, 0, 1, 1, 1350,1500, 75, Z1ROW_OFS,SL1R_OFS >(ws+FR_WL1, bl1, nullptr, 0, h); // relu, rolling row
    if (h >= 1)
      run_gemm<0, 75, 2, 4, 10, 1, 3, 10, 125,150, 0, 1, 2, 1500,3750, 10, SL1R_OFS, X2_OFS >(ws+FR_W2, b2, nullptr, h - 1, 0); // += into x2
  }

  // ---- stage 2: residual path -> d_out, then fused main path accumulates ----
  run_gemm<1,  1, 3, 3, 10, 0,  3,  5,   5, 10, 0, 3, 0, 3750,1875,375, X2_OFS,   SLR02_OFS>(ws+FR_WR02, br02, nullptr, 0, 0);
  run_gemm<0,125, 3, 3,  5, 0,  1,  5, 512,375, 0, 1, 4, 1875,   0,  5, SLR02_OFS, 0       >(ws+FR_W0C2, b0c2, gout, 0, 0);   // write d_out
#pragma unroll 1
  for (int ho = 0; ho < 2; ++ho) {
    run_gemm<0,125, 3, 3, 10, 0, 1, 10, 300,250, 0, 1, 0, 3750,3000, 10, X2_OFS,   W0ROW_OFS>(ws+FR_W0, b0, nullptr, ho, 0);
    run_gemm<1,  1, 1, 1, 10, 0, 2,  5,   5, 10, 0, 1, 1, 3000,3000,300, W0ROW_OFS,SL2_OFS  >(ws+FR_WL2, bl2, nullptr, 0, ho); // relu
  }
  run_gemm<0,300, 2, 2,  5, 0,  1,  5, 512,600, 0, 1, 6, 3000,   0,  5, SL2_OFS,   0       >(ws+FR_W0B2, b0b2, gout, 0, 0);   // += d_out
}

extern "C" void kernel_launch(void* const* d_in, const int* in_sizes, int n_in,
                              void* d_out, int out_size, void* d_ws, size_t ws_size,
                              hipStream_t stream) {
  (void)n_in; (void)out_size; (void)ws_size;
  const float* x   = (const float*)d_in[0];
  bf16_t*      ws  = (bf16_t*)d_ws;
  float*       out = (float*)d_out;

  struct PI { int widx, type, Cin, kh, K, N, off; };
  static const PI ops[15] = {
    { 1, 0,   1, 3,   3,   5, FR_C0A  },
    { 3, 1,   1, 1,  27,  18, FR_SL0  },
    { 5, 0,   5, 3,  15,  25, FR_W0B  },
    { 9, 1,   1, 1,  27,  18, FR_WR0  },
    { 7, 0,   1, 1,   1,  25, FR_W0C  },
    {19, 1,   1, 1,  18,  10, FR_WRA  },
    {17, 0,  25, 3,  75, 125, FR_WA   },
    {11, 0,  25, 2,  50,  75, FR_W1   },
    {13, 1,   1, 1,  18,  10, FR_WL1  },
    {15, 0,  75, 2, 150, 125, FR_W2   },
    {29, 1,   1, 1,  10,   5, FR_WR02 },
    {27, 0, 125, 3, 375, 512, FR_W0C2 },
    {21, 0, 125, 2, 250, 300, FR_W0   },
    {23, 1,   1, 1,  10,   5, FR_WL2  },
    {25, 0, 300, 2, 600, 512, FR_W0B2 },
  };
  for (int i = 0; i < 15; ++i) {
    int kt = (ops[i].K + 31) / 32, nt = (ops[i].N + 15) / 16;
    int total = kt * nt * 512;
    prep_frag<<<(total + NTHR - 1) / NTHR, NTHR, 0, stream>>>(
        (const float*)d_in[ops[i].widx], ws + ops[i].off,
        ops[i].type, ops[i].Cin, ops[i].kh, ops[i].K, ops[i].N, kt, nt);
  }

  int B = in_sizes[0] / 135;   // (B,1,5,27)
  msl_main<<<B / 16, NTHR, SMEM_BYTES, stream>>>(x, ws, out,
      (const float*)d_in[2],  (const float*)d_in[4],  (const float*)d_in[6],
      (const float*)d_in[10], (const float*)d_in[8],  (const float*)d_in[20],
      (const float*)d_in[18], (const float*)d_in[12], (const float*)d_in[14],
      (const float*)d_in[16], (const float*)d_in[30], (const float*)d_in[28],
      (const float*)d_in[22], (const float*)d_in[24], (const float*)d_in[26]);
}
